// ObstacleNN_82188494176425
// MI455X (gfx1250) — matrix operand order repaired and checked by reading, not compiled, not run on hardware
//
#include <hip/hip_runtime.h>
#include <hip/hip_bf16.h>

typedef _Float16 f16;
typedef __attribute__((ext_vector_type(16))) _Float16 v16h;
typedef __attribute__((ext_vector_type(8)))  float    v8f;

// ---- WMMA 16-bit fragment index maps (CDNA5 ISA 7.12.2) ----
// A (16x32 f16): lane L holds row M=L%16; half i -> K per table.
__device__ __forceinline__ int mapA_k(int lane, int i) {
    int v = i >> 1, p = i & 1;
    int grp = v >> 2, wi = v & 3;
    return grp * 16 + ((lane & 16) ? 8 : 0) + wi * 2 + p;
}
// B (32x16 f16): lane L holds col N=L%16; half i -> K per table (same as A).
__device__ __forceinline__ int mapB_k(int lane, int i) {
    return ((i & 8) ? 16 : 0) + ((lane & 16) ? 8 : 0) + (i & 7);
}

// Packed fragment layout: halves at  (((tile*KC + kc)*32 + lane)*16 + i)
// so each lane's v16h fragment is one contiguous 32-byte load.

// ---------------------------------------------------------------------------
// Pack f32 weight matrix W[Kreal x N] (row-major) into B-fragment layout.
// ---------------------------------------------------------------------------
__global__ void pack_B_frag(const float* __restrict__ W, int Kreal, int N,
                            int KC, f16* __restrict__ Bp, int total)
{
    int idx = blockIdx.x * blockDim.x + threadIdx.x;
    if (idx >= total) return;
    int i    = idx & 15;
    int lane = (idx >> 4) & 31;
    int kc   = (idx >> 9) % KC;
    int nt   = idx / (KC << 9);
    int kb   = kc * 32 + mapB_k(lane, i);
    int ncol = nt * 16 + (lane & 15);
    int kcl  = kb < Kreal ? kb : Kreal - 1;          // clamp, no divergence
    float v  = W[(size_t)kcl * N + ncol];
    Bp[idx]  = (f16)(kb < Kreal ? v : 0.0f);
}

// ---------------------------------------------------------------------------
// Pack f32 activation A[M x lda] (row-major, fully addressable up to KC*32)
// into A-fragment layout.
// ---------------------------------------------------------------------------
__global__ void pack_A_dense(const float* __restrict__ A, int lda,
                             int KC, f16* __restrict__ Ap, int total)
{
    int idx = blockIdx.x * blockDim.x + threadIdx.x;
    if (idx >= total) return;
    int i    = idx & 15;
    int lane = (idx >> 4) & 31;
    int kc   = (idx >> 9) % KC;
    int mt   = idx / (KC << 9);
    int ka   = kc * 32 + mapA_k(lane, i);
    int mrow = mt * 16 + (lane & 15);
    Ap[idx]  = (f16)A[(size_t)mrow * lda + ka];
}

// ---------------------------------------------------------------------------
// Im2col pack for 3x3 VALID conv: rows = (b,oy,ox), K = 9*Cin.  Branchless.
// ---------------------------------------------------------------------------
__global__ void pack_A_conv(const float* __restrict__ In, int H, int Wd, int Cin,
                            int outH, int outW, int Kreal, int KC, int Mreal,
                            f16* __restrict__ Ap, int total)
{
    int idx = blockIdx.x * blockDim.x + threadIdx.x;
    if (idx >= total) return;
    int i    = idx & 15;
    int lane = (idx >> 4) & 31;
    int kc   = (idx >> 9) % KC;
    int mt   = idx / (KC << 9);
    int ka   = kc * 32 + mapA_k(lane, i);
    int mrow = mt * 16 + (lane & 15);
    bool ok  = (ka < Kreal) && (mrow < Mreal);
    int kac  = ok ? ka : 0;
    int mr   = ok ? mrow : 0;
    int plane = outH * outW;
    int bb = mr / plane, pp = mr % plane;
    int oy = pp / outW,  ox = pp % outW;
    int ky = kac / (3 * Cin), rem = kac % (3 * Cin);
    int kx = rem / Cin,       ci  = rem % Cin;
    float v = In[(((size_t)bb * H + (oy + ky)) * Wd + (ox + kx)) * Cin + ci];
    Ap[idx] = (f16)(ok ? v : 0.0f);
}

// ---------------------------------------------------------------------------
// Pack dense0 input [s(6), a(2), p2 flat(6272), zeros] directly into A frags.
// ---------------------------------------------------------------------------
__global__ void pack_h0_frag(const float* __restrict__ s_cur,
                             const float* __restrict__ action, int t, int T,
                             const float* __restrict__ p2,
                             int KC, f16* __restrict__ Ap, int total)
{
    int idx = blockIdx.x * blockDim.x + threadIdx.x;
    if (idx >= total) return;
    int i    = idx & 15;
    int lane = (idx >> 4) & 31;
    int kc   = (idx >> 9) % KC;
    int mt   = idx / (KC << 9);
    int ka   = kc * 32 + mapA_k(lane, i);
    int b    = mt * 16 + (lane & 15);
    float v;
    if (ka < 6)            v = s_cur[b * 6 + ka];
    else if (ka < 8)       v = action[((size_t)b * T + t) * 2 + (ka - 6)];
    else if (ka < 6280)    v = p2[(size_t)b * 6272 + (ka - 8)];
    else                   v = 0.0f;
    Ap[idx] = (f16)v;
}

// ---------------------------------------------------------------------------
// GEMM on packed fragments: one wave per 16x16 tile.  Per k-chunk:
// two 32B loads + one v_wmma.  Bias + optional ReLU fused in epilogue.
// ---------------------------------------------------------------------------
__global__ void wmma_gemm_packed(const f16* __restrict__ Ap,
                                 const f16* __restrict__ Bp,
                                 const float* __restrict__ bias,
                                 float* __restrict__ C, int ldc,
                                 int KC, int Mreal, int relu)
{
    int lane = threadIdx.x;
    int mt = blockIdx.x, nt = blockIdx.y;
    const v16h* Av = (const v16h*)Ap + (size_t)mt * KC * 32 + lane;
    const v16h* Bv = (const v16h*)Bp + (size_t)nt * KC * 32 + lane;
    v8f acc = {};
#pragma unroll 2
    for (int kc = 0; kc < KC; ++kc) {
        v16h a = Av[(size_t)kc * 32];
        v16h b = Bv[(size_t)kc * 32];
        acc = __builtin_amdgcn_wmma_f32_16x16x32_f16(
            false, a, false, b, (short)0, acc, false, false);
    }
    int ncol  = nt * 16 + (lane & 15);
    int rbase = mt * 16 + ((lane & 16) ? 8 : 0);
    float bv = bias[ncol];
#pragma unroll
    for (int r = 0; r < 8; ++r) {
        int row = rbase + r;
        if (row < Mreal) {
            float v = acc[r] + bv;
            if (relu) v = fmaxf(v, 0.0f);
            C[(size_t)row * ldc + ncol] = v;
        }
    }
}

// ---------------------------------------------------------------------------
// 2x2 max-pool, stride 2, VALID.  NHWC.
// ---------------------------------------------------------------------------
__global__ void maxpool2(const float* __restrict__ In, float* __restrict__ Out,
                         int inW, int C, int outH, int outW, int inH, int total)
{
    int idx = blockIdx.x * blockDim.x + threadIdx.x;
    if (idx >= total) return;
    int c = idx % C;
    int t = idx / C;
    int px = t % outW; t /= outW;
    int py = t % outH;
    int b = t / outH;
    const float* base = In + (((size_t)b * inH + 2 * py) * inW + 2 * px) * C + c;
    float v = base[0];
    v = fmaxf(v, base[C]);
    v = fmaxf(v, base[(size_t)inW * C]);
    v = fmaxf(v, base[(size_t)inW * C + C]);
    Out[idx] = v;
}

// ---------------------------------------------------------------------------
// Build per-step input image (B,64,64,6): [3 raster ch, env patch, 2 action]
// ---------------------------------------------------------------------------
__global__ void build_input(const float* __restrict__ s_cur,
                            const float* __restrict__ action, int t, int T,
                            const float* __restrict__ res,
                            const float* __restrict__ env,
                            const float* __restrict__ origin,
                            float* __restrict__ z0)
{
    int b = blockIdx.x;
    const float* s = s_cur + b * 6;
    float a0 = action[((size_t)b * T + t) * 2 + 0];
    float a1 = action[((size_t)b * T + t) * 2 + 1];
    float r0 = res[(size_t)b * T];        // res[:,0,0]
    float orow = origin[b * 2 + 0];
    float ocol = origin[b * 2 + 1];
    int ccol = (int)(s[4] / 0.03f + ocol);
    int crow = (int)(s[5] / 0.03f + orow);
    float lo_r = orow - (float)crow + 32.0f;
    float lo_c = ocol - (float)ccol + 32.0f;
    int pr[3], pc[3];
    bool pv[3];
#pragma unroll
    for (int p = 0; p < 3; ++p) {
        float px = s[2 * p], py = s[2 * p + 1];
        int rr = (int)(py / r0 + lo_r);
        int cc = (int)(px / r0 + lo_c);
        pv[p] = (rr >= 0 && rr < 64 && cc >= 0 && cc < 64);
        pr[p] = rr; pc[p] = cc;
    }
    for (int pix = threadIdx.x; pix < 64 * 64; pix += blockDim.x) {
        int r = pix >> 6, c = pix & 63;
        float ch0 = (pv[0] && pr[0] == r && pc[0] == c) ? 1.0f : 0.0f;
        float ch1 = (pv[1] && pr[1] == r && pc[1] == c) ? 1.0f : 0.0f;
        float ch2 = (pv[2] && pr[2] == r && pc[2] == c) ? 1.0f : 0.0f;
        int gr = crow - 32 + r, gc = ccol - 32 + c;
        float ch3 = (gr >= 0 && gr < 256 && gc >= 0 && gc < 256)
                        ? env[((size_t)b * 256 + gr) * 256 + gc] : 0.0f;
        float* dst = z0 + ((size_t)(b * 64 + r) * 64 + c) * 6;
        dst[0] = ch0; dst[1] = ch1; dst[2] = ch2;
        dst[3] = ch3; dst[4] = a0; dst[5] = a1;
    }
}

// ---------------------------------------------------------------------------
// Final dense 256->6, residual state update, write trajectory row t+1.
// ---------------------------------------------------------------------------
__global__ void dense2_update(const float* __restrict__ h2,
                              const float* __restrict__ w2,
                              const float* __restrict__ b2,
                              float* __restrict__ s_cur,
                              float* __restrict__ out, int t)
{
    int i = blockIdx.x * blockDim.x + threadIdx.x;
    if (i >= 64 * 6) return;
    int b = i / 6, j = i % 6;
    float acc = b2[j];
    const float* h = h2 + b * 256;
    for (int k = 0; k < 256; ++k) acc += h[k] * w2[k * 6 + j];
    float sn = s_cur[i] + acc;
    s_cur[i] = sn;
    out[((size_t)b * 51 + t + 1) * 6 + j] = sn;
}

__global__ void init_state(const float* __restrict__ state,
                           float* __restrict__ s_cur,
                           float* __restrict__ out)
{
    int i = blockIdx.x * blockDim.x + threadIdx.x;
    if (i >= 64 * 6) return;
    int b = i / 6, j = i % 6;
    float v = state[((size_t)b * 51) * 6 + j];
    s_cur[i] = v;
    out[((size_t)b * 51) * 6 + j] = v;
}

extern "C" void kernel_launch(void* const* d_in, const int* in_sizes, int n_in,
                              void* d_out, int out_size, void* d_ws, size_t ws_size,
                              hipStream_t stream)
{
    (void)in_sizes; (void)n_in; (void)out_size; (void)ws_size;
    const float* action = (const float*)d_in[0];   // (64,50,2)
    const float* state  = (const float*)d_in[1];   // (64,51,6)
    const float* res    = (const float*)d_in[2];   // (64,50,1)
    const float* env    = (const float*)d_in[3];   // (64,256,256)
    const float* origin = (const float*)d_in[4];   // (64,2)
    const float* w1     = (const float*)d_in[5];   // (3,3,6,16)
    const float* b1     = (const float*)d_in[6];   // (16)
    const float* w2     = (const float*)d_in[7];   // (3,3,16,32)
    const float* b2     = (const float*)d_in[8];   // (32)
    const float* dw0    = (const float*)d_in[9];   // (6280,256)
    const float* db0    = (const float*)d_in[10];  // (256)
    const float* dw1    = (const float*)d_in[11];  // (256,256)
    const float* db1    = (const float*)d_in[12];  // (256)
    const float* dw2    = (const float*)d_in[13];  // (256,6)
    const float* db2    = (const float*)d_in[14];  // (6)
    float* out = (float*)d_out;                    // (64,51,6)

    char* ws = (char*)d_ws;
    size_t off = 0;
    auto allocf = [&](size_t nfloats) {
        float* p = (float*)(ws + off);
        off += ((nfloats * sizeof(float) + 255) & ~(size_t)255);
        return p;
    };
    auto alloch = [&](size_t nhalf) {
        f16* p = (f16*)(ws + off);
        off += ((nhalf * sizeof(f16) + 255) & ~(size_t)255);
        return p;
    };

    // f32 intermediates
    float* s_cur = allocf(64 * 6);
    float* z0  = allocf((size_t)64 * 64 * 64 * 6);    // conv1 input
    float* c1  = allocf((size_t)64 * 62 * 62 * 16);   // conv1 out
    float* p1  = allocf((size_t)64 * 31 * 31 * 16);   // pool1 out
    float* c2  = allocf((size_t)64 * 29 * 29 * 32);   // conv2 out
    float* p2  = allocf((size_t)64 * 14 * 14 * 32);   // pool2 out
    float* h1v = allocf((size_t)64 * 256);
    float* h2v = allocf((size_t)64 * 256);

    // GEMM tile geometry
    const int MT_C1 = 15376, KC_C1 = 2;   // conv1: M=246016, K=54->64
    const int MT_C2 = 3364,  KC_C2 = 5;   // conv2: M=53824,  K=144->160
    const int MT_D  = 4;                  // dense M=64
    const int KC_D0 = 197, KC_D1 = 8;     // K=6280->6304, K=256
    const int NT_C1 = 1, NT_C2 = 2, NT_D = 16;

    // packed f16 fragments
    f16* ApC1 = alloch((size_t)MT_C1 * KC_C1 * 512);
    f16* ApC2 = alloch((size_t)MT_C2 * KC_C2 * 512);
    f16* Ap0  = alloch((size_t)MT_D  * KC_D0 * 512);
    f16* Ap1  = alloch((size_t)MT_D  * KC_D1 * 512);
    f16* BpW1 = alloch((size_t)NT_C1 * KC_C1 * 512);
    f16* BpW2 = alloch((size_t)NT_C2 * KC_C2 * 512);
    f16* BpD0 = alloch((size_t)NT_D  * KC_D0 * 512);
    f16* BpD1 = alloch((size_t)NT_D  * KC_D1 * 512);

    auto cdiv = [](int a, int b) { return (a + b - 1) / b; };

    // ---- pack weights once per launch ----
    {
        int n;
        n = NT_C1 * KC_C1 * 512;
        pack_B_frag<<<cdiv(n, 256), 256, 0, stream>>>(w1, 54, 16, KC_C1, BpW1, n);
        n = NT_C2 * KC_C2 * 512;
        pack_B_frag<<<cdiv(n, 256), 256, 0, stream>>>(w2, 144, 32, KC_C2, BpW2, n);
        n = NT_D * KC_D0 * 512;
        pack_B_frag<<<cdiv(n, 256), 256, 0, stream>>>(dw0, 6280, 256, KC_D0, BpD0, n);
        n = NT_D * KC_D1 * 512;
        pack_B_frag<<<cdiv(n, 256), 256, 0, stream>>>(dw1, 256, 256, KC_D1, BpD1, n);
    }

    init_state<<<2, 192, 0, stream>>>(state, s_cur, out);

    const int T = 50;
    for (int t = 0; t < T; ++t) {
        build_input<<<64, 256, 0, stream>>>(s_cur, action, t, T, res, env, origin, z0);

        // conv1: im2col pack + GEMM (M=246016, N=16, K=54)
        {
            int n = MT_C1 * KC_C1 * 512;
            pack_A_conv<<<cdiv(n, 256), 256, 0, stream>>>(
                z0, 64, 64, 6, 62, 62, 54, KC_C1, 64 * 62 * 62, ApC1, n);
            wmma_gemm_packed<<<dim3(MT_C1, NT_C1), 32, 0, stream>>>(
                ApC1, BpW1, b1, c1, 16, KC_C1, 64 * 62 * 62, 1);
        }
        maxpool2<<<cdiv(64 * 31 * 31 * 16, 256), 256, 0, stream>>>(
            c1, p1, 62, 16, 31, 31, 62, 64 * 31 * 31 * 16);

        // conv2: im2col pack + GEMM (M=53824, N=32, K=144)
        {
            int n = MT_C2 * KC_C2 * 512;
            pack_A_conv<<<cdiv(n, 256), 256, 0, stream>>>(
                p1, 31, 31, 16, 29, 29, 144, KC_C2, 64 * 29 * 29, ApC2, n);
            wmma_gemm_packed<<<dim3(MT_C2, NT_C2), 32, 0, stream>>>(
                ApC2, BpW2, b2, c2, 32, KC_C2, 64 * 29 * 29, 1);
        }
        maxpool2<<<cdiv(64 * 14 * 14 * 32, 256), 256, 0, stream>>>(
            c2, p2, 29, 32, 14, 14, 29, 64 * 14 * 14 * 32);

        // dense0: pack [s,a,flat] + GEMM (64 x 6280 x 256), relu
        {
            int n = MT_D * KC_D0 * 512;
            pack_h0_frag<<<cdiv(n, 256), 256, 0, stream>>>(
                s_cur, action, t, T, p2, KC_D0, Ap0, n);
            wmma_gemm_packed<<<dim3(MT_D, NT_D), 32, 0, stream>>>(
                Ap0, BpD0, db0, h1v, 256, KC_D0, 64, 1);
        }
        // dense1: pack h1 + GEMM (64 x 256 x 256), relu
        {
            int n = MT_D * KC_D1 * 512;
            pack_A_dense<<<cdiv(n, 256), 256, 0, stream>>>(h1v, 256, KC_D1, Ap1, n);
            wmma_gemm_packed<<<dim3(MT_D, NT_D), 32, 0, stream>>>(
                Ap1, BpD1, db1, h2v, 256, KC_D1, 64, 1);
        }

        dense2_update<<<2, 192, 0, stream>>>(h2v, dw2, db2, s_cur, out, t);
    }
}
